// PixelCrossAttentionRefiner_51960514347049
// MI455X (gfx1250) — matrix operand order repaired and checked by reading, not compiled, not run on hardware
//
#include <hip/hip_runtime.h>

typedef _Float16 f16;
typedef __attribute__((ext_vector_type(16))) _Float16 v16h;
typedef __attribute__((ext_vector_type(8)))  float    v8f;

#define NN 16384   // H*W
// packed weight-fragment bases (each fragment = 512 halves = 32 lanes * 16)
#define FRAG_Q1 0
#define FRAG_K1 12
#define FRAG_V1 24
#define FRAG_OUT 32
#define FRAG_Q2 40
#define FRAG_K2 60
#define FRAG_V2 80
#define NFRAGS 96
#define WFRAG_HALVES (NFRAGS * 512)
#define BIAS_OFF_BYTES (WFRAG_HALVES * 2)   // 98304
#define ZWS_OFF_BYTES  131072               // 128KB aligned; z buffer [B,64,NN] f32

__device__ __forceinline__ v8f wmma_acc(v8f c, v16h a, v16h b) {
  // D = A(16x32 f16) * B(32x16 f16) + C(f32)
  return __builtin_amdgcn_wmma_f32_16x16x32_f16(false, a, false, b, (short)0, c, false, false);
}

__device__ __forceinline__ v16h load_wfrag(const f16* __restrict__ wf, int frag, int lane) {
  return *(const v16h*)(wf + ((size_t)frag << 9) + ((size_t)lane << 4));
}

// A-fragment (16x32 f16) layout: lane row m=lane&15, hi=lane>>4,
// half i -> K = (i&8)*2 + hi*8 + (i&7)
template<int KSTEPS, typename F>
__device__ __forceinline__ void build_a(v16h (&a)[KSTEPS], F f, int hi) {
  #pragma unroll
  for (int ks = 0; ks < KSTEPS; ++ks) {
    #pragma unroll
    for (int i = 0; i < 16; ++i) {
      int k = ks * 32 + ((i & 8) << 1) + hi * 8 + (i & 7);
      a[ks][i] = (f16)f(k);
    }
  }
}

// one 16x64 output tile: accumulate KSTEPS WMMAs per 16-wide N tile, add bias,
// write to LDS buffer dst[row*64+col] (T = float or f16).
template<int KSTEPS, typename T>
__device__ __forceinline__ void gemm_tile(const v16h (&a)[KSTEPS],
                                          const f16* __restrict__ wf, int fragBase,
                                          const float* __restrict__ biases, int bidx,
                                          T* dst, int m, int hi, int lane) {
  #pragma unroll
  for (int nt = 0; nt < 4; ++nt) {
    v8f c = {};
    #pragma unroll
    for (int ks = 0; ks < KSTEPS; ++ks)
      c = wmma_acc(c, a[ks], load_wfrag(wf, fragBase + ks * 4 + nt, lane));
    float bb = biases[bidx * 64 + nt * 16 + m];
    // C/D layout: lane col = nt*16 + (lane&15); VGPR v -> row v + hi*8
    #pragma unroll
    for (int v = 0; v < 8; ++v)
      dst[(v + hi * 8) * 64 + nt * 16 + m] = (T)(c[v] + bb);
  }
}

// ---------------------------------------------------------------------------
// Weight-folding prep kernel: Wc[k,n] = sum_e projW[e,k] * in_proj_w[sel*64+n,e]
// packed into WMMA B-fragment layout (32x16 f16): lane col n=lane&15,
// half i -> K = kbase + (i&8)*2 + (lane>>4)*8 + (i&7).
// ---------------------------------------------------------------------------
__global__ void prep_weights_kernel(
    const float* __restrict__ q1_w, const float* __restrict__ k1_w,
    const float* __restrict__ v1_w, const float* __restrict__ q2_w,
    const float* __restrict__ k2_w, const float* __restrict__ v2_w,
    const float* __restrict__ out_w,
    const float* __restrict__ q1_b, const float* __restrict__ k1_b,
    const float* __restrict__ v1_b, const float* __restrict__ q2_b,
    const float* __restrict__ k2_b, const float* __restrict__ v2_b,
    const float* __restrict__ out_b,
    const float* __restrict__ in_proj_w, const float* __restrict__ in_proj_b,
    f16* __restrict__ wfrags, float* __restrict__ biases)
{
  int gid = blockIdx.x * blockDim.x + threadIdx.x;
  if (gid < WFRAG_HALVES) {
    int frag = gid >> 9;
    int e    = gid & 511;
    int lane = e >> 4;
    int i    = e & 15;
    int hi   = lane >> 4;
    int mat, base;
    if      (frag < 12) { mat = 0; base = FRAG_Q1; }
    else if (frag < 24) { mat = 1; base = FRAG_K1; }
    else if (frag < 32) { mat = 2; base = FRAG_V1; }
    else if (frag < 40) { mat = 3; base = FRAG_OUT; }
    else if (frag < 60) { mat = 4; base = FRAG_Q2; }
    else if (frag < 80) { mat = 5; base = FRAG_K2; }
    else                { mat = 6; base = FRAG_V2; }
    int local = frag - base;
    int ks = local >> 2;
    int nt = local & 3;
    int K  = ks * 32 + ((i & 8) << 1) + hi * 8 + (i & 7);      // B-fragment K mapping
    int n  = nt * 16 + (lane & 15);
    float val = 0.0f;
    if (mat == 3) {
      val = out_w[n * 64 + K];           // K < 64 always here
    } else {
      const float* srcW; int ld, Kact, sel;
      switch (mat) {
        case 0: srcW = q1_w; ld = 66;  Kact = 66;  sel = 0; break;
        case 1: srcW = k1_w; ld = 66;  Kact = 66;  sel = 1; break;
        case 2: srcW = v1_w; ld = 64;  Kact = 64;  sel = 2; break;
        case 4: srcW = q2_w; ld = 130; Kact = 130; sel = 0; break;
        case 5: srcW = k2_w; ld = 130; Kact = 130; sel = 1; break;
        default: srcW = v2_w; ld = 128; Kact = 128; sel = 2; break;
      }
      if (K < Kact) {
        const float* wrow = in_proj_w + (size_t)(sel * 64 + n) * 64;
        float s = 0.0f;
        for (int ee = 0; ee < 64; ++ee) s += srcW[(size_t)ee * ld + K] * wrow[ee];
        val = s;
      }
    }
    wfrags[gid] = (f16)val;
  }
  int bid = gid - WFRAG_HALVES;
  if (bid >= 0 && bid < 7 * 64) {
    int which = bid >> 6;
    int n     = bid & 63;
    float v;
    if (which == 3) {
      v = out_b[n];
    } else {
      const float* bsrc; int sel;
      switch (which) {
        case 0: bsrc = q1_b; sel = 0; break;
        case 1: bsrc = k1_b; sel = 1; break;
        case 2: bsrc = v1_b; sel = 2; break;
        case 4: bsrc = q2_b; sel = 0; break;
        case 5: bsrc = k2_b; sel = 1; break;
        default: bsrc = v2_b; sel = 2; break;
      }
      const float* wrow = in_proj_w + (size_t)(sel * 64 + n) * 64;
      float s = in_proj_b[sel * 64 + n];
      for (int ee = 0; ee < 64; ++ee) s += bsrc[ee] * wrow[ee];
      v = s;
    }
    biases[bid] = v;
  }
}

// ---------------------------------------------------------------------------
// Fused stage kernel: 1 wave = 16 query pixels (one row segment).
// Candidates for these pixels are 128 CONTIGUOUS scrambled positions
// (torch-reshape semantics: row (p,kk) -> flat p*8+kk over (K,H,W)).
// STAGE==1: dest = z workspace [B,64,NN]. STAGE==2: dest = output [B,64,H,W].
// ---------------------------------------------------------------------------
template<int STAGE>
__global__ __launch_bounds__(32) void refine_stage_kernel(
    const float* __restrict__ HF_star, const float* __restrict__ HF_cands,
    const float* __restrict__ Z_cands, const f16* __restrict__ wfrags,
    const float* __restrict__ biases, const float* __restrict__ z_in,
    float* __restrict__ dest)
{
  __shared__ float s_qh[16 * 64];   // folded q-projection (f32)
  __shared__ f16   s_kh[128 * 64];  // folded k-projection, 128 candidate rows
  __shared__ f16   s_vh[128 * 64];  // folded v-projection
  __shared__ f16   s_z [16 * 64];   // attention output (A of out-proj)
  __shared__ float s_o [16 * 64];   // out-proj result (f32, for transposed store)

  constexpr int KSQ = (STAGE == 1) ? 3 : 5;   // 96 / 160 padded K
  constexpr int KSK = (STAGE == 1) ? 3 : 5;
  constexpr int KSV = (STAGE == 1) ? 2 : 4;   // 64 / 128
  constexpr int FQ  = (STAGE == 1) ? FRAG_Q1 : FRAG_Q2;
  constexpr int FK  = (STAGE == 1) ? FRAG_K1 : FRAG_K2;
  constexpr int FV  = (STAGE == 1) ? FRAG_V1 : FRAG_V2;
  constexpr int BQ  = (STAGE == 1) ? 0 : 4;
  constexpr int BK  = (STAGE == 1) ? 1 : 5;
  constexpr int BV  = (STAGE == 1) ? 2 : 6;

  const int lane = threadIdx.x;
  const int m    = lane & 15;
  const int hi   = lane >> 4;
  const int tile = blockIdx.x;          // 4096 tiles
  const int b     = tile >> 10;         // 1024 tiles per batch
  const int pixq0 = (tile & 1023) * 16;
  const int row   = pixq0 >> 7;
  const int col0  = pixq0 & 127;
  const int pixq  = pixq0 + m;          // this lane's query pixel

  const float ciq = (float)row * (1.0f / 127.0f);
  const float cjq = (float)(col0 + m) * (1.0f / 127.0f);

  // scrambled candidate block: 128 contiguous positions in one candidate plane
  const int spBase = (pixq0 * 8) & (NN - 1);
  const int kIdx   = (pixq0 * 8) >> 14;

  const float* qbase  = HF_star + (size_t)b * 64 * NN + pixq;                     // [c*NN]
  const float* kcbase = HF_cands + ((size_t)(b * 8 + kIdx)) * 64 * NN + spBase;   // [+crow +c*NN]
  const float* vcbase = Z_cands  + ((size_t)(b * 8 + kIdx)) * 64 * NN + spBase;
  const float* zq = z_in + (size_t)b * 64 * NN + pixq;                            // [e*NN]
  const float* zc = z_in + (size_t)b * 64 * NN + spBase;                          // [+crow +e*NN]

  // ---- folded Q projection: qh[16,64] -> s_qh (f32) ----
  {
    v16h a[KSQ];
    build_a<KSQ>(a, [&](int k) -> float {
      if (k < 64) return qbase[(size_t)k * NN];
      if (STAGE == 1) {
        if (k == 64) return ciq;
        if (k == 65) return cjq;
        return 0.0f;
      } else {
        if (k < 128) return zq[(size_t)(k - 64) * NN];
        if (k == 128) return ciq;
        if (k == 129) return cjq;
        return 0.0f;
      }
    }, hi);
    gemm_tile<KSQ>(a, wfrags, FQ, biases, BQ, s_qh, m, hi, lane);
  }

  // ---- folded K projection: kh[128,64] -> s_kh (f16) ----
  for (int t = 0; t < 8; ++t) {
    const int crow = t * 16 + m;
    const int sp   = spBase + crow;
    const float cic = (float)(sp >> 7) * (1.0f / 127.0f);
    const float cjc = (float)(sp & 127) * (1.0f / 127.0f);
    v16h a[KSK];
    build_a<KSK>(a, [&](int k) -> float {
      if (k < 64) return kcbase[crow + (size_t)k * NN];
      if (STAGE == 1) {
        if (k == 64) return cic;
        if (k == 65) return cjc;
        return 0.0f;
      } else {
        if (k < 128) return zc[crow + (size_t)(k - 64) * NN];
        if (k == 128) return cic;
        if (k == 129) return cjc;
        return 0.0f;
      }
    }, hi);
    gemm_tile<KSK>(a, wfrags, FK, biases, BK, s_kh + t * 16 * 64, m, hi, lane);
  }

  // ---- folded V projection: vh[128,64] -> s_vh (f16) ----
  for (int t = 0; t < 8; ++t) {
    const int crow = t * 16 + m;
    v16h a[KSV];
    build_a<KSV>(a, [&](int k) -> float {
      if (k < 64) return vcbase[crow + (size_t)k * NN];
      if (STAGE == 2 && k < 128) return zc[crow + (size_t)(k - 64) * NN];
      return 0.0f;
    }, hi);
    gemm_tile<KSV>(a, wfrags, FV, biases, BV, s_vh + t * 16 * 64, m, hi, lane);
  }

  __syncthreads();

  // ---- attention: each lane handles pixel m for 2 heads (hi*2, hi*2+1) ----
  #pragma unroll
  for (int hh = 0; hh < 2; ++hh) {
    const int h = hi * 2 + hh;
    float q[16];
    #pragma unroll
    for (int d = 0; d < 16; ++d) q[d] = s_qh[m * 64 + h * 16 + d];
    float sc[8];
    float mx = -1e30f;
    #pragma unroll
    for (int kk = 0; kk < 8; ++kk) {
      const f16* kr = s_kh + (m * 8 + kk) * 64 + h * 16;  // scrambled row p*8+kk
      float s = 0.0f;
      #pragma unroll
      for (int d = 0; d < 16; ++d) s += q[d] * (float)kr[d];
      s *= 0.25f;                                          // 1/sqrt(16)
      sc[kk] = s;
      mx = fmaxf(mx, s);
    }
    float den = 0.0f;
    #pragma unroll
    for (int kk = 0; kk < 8; ++kk) { sc[kk] = __expf(sc[kk] - mx); den += sc[kk]; }
    const float inv = 1.0f / den;
    float o[16];
    #pragma unroll
    for (int d = 0; d < 16; ++d) o[d] = 0.0f;
    #pragma unroll
    for (int kk = 0; kk < 8; ++kk) {
      const float w = sc[kk] * inv;
      const f16* vr = s_vh + (m * 8 + kk) * 64 + h * 16;
      #pragma unroll
      for (int d = 0; d < 16; ++d) o[d] += w * (float)vr[d];
    }
    #pragma unroll
    for (int d = 0; d < 16; ++d) s_z[m * 64 + h * 16 + d] = (f16)o[d];
  }

  __syncthreads();

  // ---- output projection: z[16,64] @ out_w^T + out_b -> s_o (f32) ----
  {
    v16h a[2];
    #pragma unroll
    for (int ks = 0; ks < 2; ++ks)
      #pragma unroll
      for (int i = 0; i < 16; ++i)
        a[ks][i] = s_z[m * 64 + ks * 32 + ((i & 8) << 1) + hi * 8 + (i & 7)];
    gemm_tile<2>(a, wfrags, FRAG_OUT, biases, 3, s_o, m, hi, lane);
  }

  __syncthreads();

  // ---- transposed coalesced store: dest[(b*64+e)*NN + pixq0 + m] ----
  float* dbase = dest + (size_t)b * 64 * NN + pixq0 + m;
  #pragma unroll
  for (int e2 = 0; e2 < 32; ++e2) {
    const int e = e2 * 2 + hi;
    dbase[(size_t)e * NN] = s_o[m * 64 + e];
  }
}

extern "C" void kernel_launch(void* const* d_in, const int* in_sizes, int n_in,
                              void* d_out, int out_size, void* d_ws, size_t ws_size,
                              hipStream_t stream) {
  const float* HF_star   = (const float*)d_in[0];
  const float* HF_cands  = (const float*)d_in[1];
  const float* Z_cands   = (const float*)d_in[2];
  const float* q1_w = (const float*)d_in[3];  const float* q1_b = (const float*)d_in[4];
  const float* k1_w = (const float*)d_in[5];  const float* k1_b = (const float*)d_in[6];
  const float* v1_w = (const float*)d_in[7];  const float* v1_b = (const float*)d_in[8];
  const float* q2_w = (const float*)d_in[9];  const float* q2_b = (const float*)d_in[10];
  const float* k2_w = (const float*)d_in[11]; const float* k2_b = (const float*)d_in[12];
  const float* v2_w = (const float*)d_in[13]; const float* v2_b = (const float*)d_in[14];
  const float* in_proj_w = (const float*)d_in[15];
  const float* in_proj_b = (const float*)d_in[16];
  const float* out_w = (const float*)d_in[17];
  const float* out_b = (const float*)d_in[18];

  f16*   wfrags = (f16*)d_ws;
  float* biases = (float*)((char*)d_ws + BIAS_OFF_BYTES);
  float* z_ws   = (float*)((char*)d_ws + ZWS_OFF_BYTES);   // [B,64,NN] f32, 16 MB
  float* out    = (float*)d_out;

  // 1) fold (projection x in_proj) weight products into packed f16 B-fragments
  const int prepTotal = WFRAG_HALVES + 7 * 64;
  prep_weights_kernel<<<(prepTotal + 255) / 256, 256, 0, stream>>>(
      q1_w, k1_w, v1_w, q2_w, k2_w, v2_w, out_w,
      q1_b, k1_b, v1_b, q2_b, k2_b, v2_b, out_b,
      in_proj_w, in_proj_b, wfrags, biases);

  // 2) stage 1: z_star -> z_ws ; 3) stage 2: z_hat -> out
  const int nTiles = 4 * NN / 16;  // 4096
  refine_stage_kernel<1><<<nTiles, 32, 0, stream>>>(
      HF_star, HF_cands, Z_cands, wfrags, biases, z_ws /*unused*/, z_ws);
  refine_stage_kernel<2><<<nTiles, 32, 0, stream>>>(
      HF_star, HF_cands, Z_cands, wfrags, biases, z_ws, out);
}
